// MultiModalBartDecoder_span_11458972745739
// MI455X (gfx1250) — matrix operand order repaired and checked by reading, not compiled, not run on hardware
//
#include <hip/hip_runtime.h>
#include <hip/hip_bf16.h>

#define BATCH     48
#define LP1       385
#define LSEQ      384
#define NSRC      256
#define HDIM      768
#define IMG_PREF  51
#define NENC      307
#define OUTC      262
#define SRC_START 6
#define PAD_ID    1
#define EOS_ID    2
#define LBL_START 50260

#define MROWS     (BATCH * NSRC)   // 12288
#define HROWS     (BATCH * LSEQ)   // 18432

typedef __attribute__((ext_vector_type(16))) __bf16 v16bf;
typedef __attribute__((ext_vector_type(8)))  __bf16 v8bf;
typedef __attribute__((ext_vector_type(8)))  float  v8f;

// WMMA fragment loaders (wave32, v_wmma_f32_16x16x32_bf16, ISA 7.12.2)
__device__ __forceinline__ v16bf load_a_bf16(const __bf16* rowk0, int h) {
    v8bf lo = *(const v8bf*)(rowk0 + 16 * h);
    v8bf hi = *(const v8bf*)(rowk0 + 16 * h + 8);
    v16bf a;
#pragma unroll
    for (int i = 0; i < 8; ++i) { a[i] = lo[i]; a[8 + i] = hi[i]; }
    return a;
}

__device__ __forceinline__ v16bf load_a_f32(const float* rowk0, int h) {
    const float* p0 = rowk0 + 16 * h;
    const float* p1 = rowk0 + 16 * h + 8;
    v16bf a;
#pragma unroll
    for (int i = 0; i < 8; ++i) {
        a[i]     = (__bf16)p0[i];
        a[8 + i] = (__bf16)p1[i];
    }
    return a;
}

__device__ __forceinline__ v16bf load_b_rowNK(const __bf16* rowk0, int h) {
    return *(const v16bf*)(rowk0 + 16 * h);
}

__device__ __forceinline__ v8f wmma_bf16(v16bf a, v16bf b, v8f c) {
    return __builtin_amdgcn_wmma_f32_16x16x32_bf16(false, a, false, b,
                                                   (short)0, c, false, false);
}

__global__ void wconv_kernel(const float* __restrict__ W1,
                             const float* __restrict__ W2,
                             __bf16* __restrict__ W1t,
                             __bf16* __restrict__ W2t) {
    int idx = blockIdx.x * blockDim.x + threadIdx.x;
    if (idx >= HDIM * HDIM) return;
    int n = idx / HDIM;
    int k = idx % HDIM;
    W1t[idx] = (__bf16)W1[k * HDIM + n];
    W2t[idx] = (__bf16)W2[k * HDIM + n];
}

__global__ void prep_kernel(const int* __restrict__ tokens,
                            const int* __restrict__ src_tokens,
                            const int* __restrict__ enc_mask,
                            int* __restrict__ tok,
                            int* __restrict__ maskArr) {
    int b = blockIdx.x;
    int tid = threadIdx.x;
    __shared__ int s_firstPad, s_firstEos;
    if (tid == 0) { s_firstPad = LP1; s_firstEos = NSRC; }
    __syncthreads();
    for (int i = tid; i < LP1; i += blockDim.x)
        if (tokens[b * LP1 + i] == PAD_ID) atomicMin(&s_firstPad, i);
    for (int i = tid; i < NSRC; i += blockDim.x)
        if (src_tokens[b * NSRC + i] == EOS_ID) atomicMin(&s_firstEos, i);
    __syncthreads();
    int fp = s_firstPad;
    int fe = s_firstEos;
    const int lut[SRC_START] = {0, 2, 50260, 50261, 50262, 50263};
    for (int l = tid; l < LSEQ; l += blockDim.x) {
        int t = tokens[b * LP1 + l];
        int v;
        if (t < SRC_START) {
            v = lut[t];
        } else {
            int si = t - SRC_START;
            si = si < 0 ? 0 : (si > NSRC - 1 ? NSRC - 1 : si);
            v = src_tokens[b * NSRC + si];
        }
        if (l > fp) v = PAD_ID;
        tok[b * LSEQ + l] = v;
    }
    for (int n = tid; n < NSRC; n += blockDim.x) {
        bool m = (enc_mask[b * NENC + IMG_PREF + n] == 0) || (n >= fe);
        maskArr[b * NSRC + n] = m ? 1 : 0;
    }
}

__global__ void hidden_kernel(const int* __restrict__ tok,
                              const float* __restrict__ embed,
                              __bf16* __restrict__ Hmat) {
    int idx = blockIdx.x * blockDim.x + threadIdx.x;
    if (idx >= HROWS * (HDIM / 8)) return;
    int row = idx / (HDIM / 8);
    int c0  = (idx % (HDIM / 8)) * 8;
    int t = tok[row];
    const float* src = embed + (size_t)t * HDIM + c0;
    v8bf v;
#pragma unroll
    for (int i = 0; i < 8; ++i) v[i] = (__bf16)src[i];
    *(v8bf*)(Hmat + (size_t)row * HDIM + c0) = v;
}

// GEMM1: H1 = relu(enc_slice @ W1 + b1), 32x32 tile/wave (4 WMMA per k-step)
__global__ __launch_bounds__(32)
void gemm1_kernel(const float* __restrict__ enc,
                  const __bf16* __restrict__ W1t,
                  const float* __restrict__ b1,
                  __bf16* __restrict__ H1) {
    int n0 = blockIdx.x * 32;
    int m0 = blockIdx.y * 32;
    int lane = threadIdx.x;
    int ml = lane & 15, h = lane >> 4;

    int row0 = m0 + ml, row1 = m0 + 16 + ml;
    const float* arow0 = enc + ((size_t)((row0 >> 8) * NENC + IMG_PREF + (row0 & 255))) * HDIM;
    const float* arow1 = enc + ((size_t)((row1 >> 8) * NENC + IMG_PREF + (row1 & 255))) * HDIM;
    const __bf16* brow0 = W1t + (size_t)(n0 + ml) * HDIM;
    const __bf16* brow1 = W1t + (size_t)(n0 + 16 + ml) * HDIM;

    v8f c00 = {}, c01 = {}, c10 = {}, c11 = {};
    for (int k0 = 0; k0 < HDIM; k0 += 32) {
        v16bf a0 = load_a_f32(arow0 + k0, h);
        v16bf a1 = load_a_f32(arow1 + k0, h);
        v16bf b0 = load_b_rowNK(brow0 + k0, h);
        v16bf b1v = load_b_rowNK(brow1 + k0, h);
        c00 = wmma_bf16(a0, b0, c00);
        c01 = wmma_bf16(a0, b1v, c01);
        c10 = wmma_bf16(a1, b0, c10);
        c11 = wmma_bf16(a1, b1v, c11);
    }
    int nc0 = n0 + ml, nc1 = n0 + 16 + ml;
    float bi0 = b1[nc0], bi1 = b1[nc1];
#pragma unroll
    for (int r = 0; r < 8; ++r) {
        int ma = m0 + 8 * h + r;
        int mb = m0 + 16 + 8 * h + r;
        float y;
        y = c00[r] + bi0; H1[(size_t)ma * HDIM + nc0] = (__bf16)(y > 0.f ? y : 0.f);
        y = c01[r] + bi1; H1[(size_t)ma * HDIM + nc1] = (__bf16)(y > 0.f ? y : 0.f);
        y = c10[r] + bi0; H1[(size_t)mb * HDIM + nc0] = (__bf16)(y > 0.f ? y : 0.f);
        y = c11[r] + bi1; H1[(size_t)mb * HDIM + nc1] = (__bf16)(y > 0.f ? y : 0.f);
    }
}

// GEMM2: K = 0.5*(H1 @ W2 + b2 + embed[src_tokens]), 32x32 tile/wave
__global__ __launch_bounds__(32)
void gemm2_kernel(const __bf16* __restrict__ H1,
                  const __bf16* __restrict__ W2t,
                  const float* __restrict__ b2,
                  const int* __restrict__ src_tokens,
                  const float* __restrict__ embed,
                  __bf16* __restrict__ Kmat) {
    int n0 = blockIdx.x * 32;
    int m0 = blockIdx.y * 32;
    int lane = threadIdx.x;
    int ml = lane & 15, h = lane >> 4;

    const __bf16* arow0 = H1 + (size_t)(m0 + ml) * HDIM;
    const __bf16* arow1 = H1 + (size_t)(m0 + 16 + ml) * HDIM;
    const __bf16* brow0 = W2t + (size_t)(n0 + ml) * HDIM;
    const __bf16* brow1 = W2t + (size_t)(n0 + 16 + ml) * HDIM;

    v8f c00 = {}, c01 = {}, c10 = {}, c11 = {};
    for (int k0 = 0; k0 < HDIM; k0 += 32) {
        v16bf a0 = load_a_bf16(arow0 + k0, h);
        v16bf a1 = load_a_bf16(arow1 + k0, h);
        v16bf b0 = load_b_rowNK(brow0 + k0, h);
        v16bf b1v = load_b_rowNK(brow1 + k0, h);
        c00 = wmma_bf16(a0, b0, c00);
        c01 = wmma_bf16(a0, b1v, c01);
        c10 = wmma_bf16(a1, b0, c10);
        c11 = wmma_bf16(a1, b1v, c11);
    }
    int nc0 = n0 + ml, nc1 = n0 + 16 + ml;
    float bi0 = b2[nc0], bi1 = b2[nc1];
#pragma unroll
    for (int r = 0; r < 8; ++r) {
        int ma = m0 + 8 * h + r;
        int mb = m0 + 16 + 8 * h + r;
        int sta = src_tokens[(ma >> 8) * NSRC + (ma & 255)];
        int stb = src_tokens[(mb >> 8) * NSRC + (mb & 255)];
        Kmat[(size_t)ma * HDIM + nc0] =
            (__bf16)(0.5f * (c00[r] + bi0 + embed[(size_t)sta * HDIM + nc0]));
        Kmat[(size_t)ma * HDIM + nc1] =
            (__bf16)(0.5f * (c01[r] + bi1 + embed[(size_t)sta * HDIM + nc1]));
        Kmat[(size_t)mb * HDIM + nc0] =
            (__bf16)(0.5f * (c10[r] + bi0 + embed[(size_t)stb * HDIM + nc0]));
        Kmat[(size_t)mb * HDIM + nc1] =
            (__bf16)(0.5f * (c11[r] + bi1 + embed[(size_t)stb * HDIM + nc1]));
    }
}

// Score GEMM: 4-wave block, 64x64 tile; B tile staged via async LDS copies.
__global__ __launch_bounds__(128)
void score_kernel(const __bf16* __restrict__ Hmat,
                  const __bf16* __restrict__ Kmat,
                  const int* __restrict__ maskArr,
                  float* __restrict__ out) {
    int b   = blockIdx.z;
    int n0b = blockIdx.x * 64;
    int m0b = blockIdx.y * 64;
    int tid  = threadIdx.x;
    int w    = tid >> 5;
    int lane = tid & 31;
    int ml = lane & 15, h = lane >> 4;

    __shared__ __bf16 sB[64 * 32];   // [n_row][k], 4 KB

    const __bf16* arow  = Hmat + ((size_t)b * LSEQ + m0b + w * 16 + ml) * HDIM;
    const __bf16* Bbase = Kmat + ((size_t)b * NSRC + n0b) * HDIM;
    unsigned int ldsBase = (unsigned int)(uintptr_t)(&sB[0]);

    v8f c0 = {}, c1 = {}, c2 = {}, c3 = {};
    for (int k0 = 0; k0 < HDIM; k0 += 32) {
#pragma unroll
        for (int cc = 0; cc < 2; ++cc) {
            int c   = tid + cc * 128;
            int row = c >> 2;
            int seg = c & 3;
            const __bf16* gp = Bbase + (size_t)row * HDIM + k0 + seg * 8;
            unsigned int ld = ldsBase + (unsigned int)(row * 64 + seg * 16);
            asm volatile("global_load_async_to_lds_b128 %0, %1, off"
                         :: "v"(ld), "v"((unsigned long long)(uintptr_t)gp)
                         : "memory");
        }
        asm volatile("s_wait_asynccnt 0x0" ::: "memory");
        __syncthreads();

        v16bf a = load_a_bf16(arow + k0, h);
        const __bf16* sb = &sB[0];
        v16bf b0 = *(const v16bf*)(sb + (0 * 16 + ml) * 32 + 16 * h);
        v16bf b1 = *(const v16bf*)(sb + (1 * 16 + ml) * 32 + 16 * h);
        v16bf b2 = *(const v16bf*)(sb + (2 * 16 + ml) * 32 + 16 * h);
        v16bf b3 = *(const v16bf*)(sb + (3 * 16 + ml) * 32 + 16 * h);
        c0 = wmma_bf16(a, b0, c0);
        c1 = wmma_bf16(a, b1, c1);
        c2 = wmma_bf16(a, b2, c2);
        c3 = wmma_bf16(a, b3, c3);
        __syncthreads();
    }

    bool m0k = maskArr[b * NSRC + n0b +  0 + ml] != 0;
    bool m1k = maskArr[b * NSRC + n0b + 16 + ml] != 0;
    bool m2k = maskArr[b * NSRC + n0b + 32 + ml] != 0;
    bool m3k = maskArr[b * NSRC + n0b + 48 + ml] != 0;
    int lbase = m0b + w * 16 + 8 * h;
#pragma unroll
    for (int r = 0; r < 8; ++r) {
        float* orow = out + ((size_t)(b * LSEQ + lbase + r)) * OUTC + SRC_START + n0b + ml;
        orow[ 0] = m0k ? -1e32f : c0[r];
        orow[16] = m1k ? -1e32f : c1[r];
        orow[32] = m2k ? -1e32f : c2[r];
        orow[48] = m3k ? -1e32f : c3[r];
    }
}

__global__ void tageos_kernel(const __bf16* __restrict__ Hmat,
                              const float* __restrict__ embed,
                              float* __restrict__ out) {
    int wid  = threadIdx.x >> 5;
    int lane = threadIdx.x & 31;
    int row  = blockIdx.x * (blockDim.x >> 5) + wid;
    if (row >= HROWS) return;
    const __bf16* hrow = Hmat + (size_t)row * HDIM;
    const float* t0 = embed + (size_t)(LBL_START + 0) * HDIM;
    const float* t1 = embed + (size_t)(LBL_START + 1) * HDIM;
    const float* t2 = embed + (size_t)(LBL_START + 2) * HDIM;
    const float* te = embed + (size_t)EOS_ID * HDIM;
    float a0 = 0.f, a1 = 0.f, a2 = 0.f, ae = 0.f;
    for (int k = lane; k < HDIM; k += 32) {
        float hv = (float)hrow[k];
        a0 += hv * t0[k];
        a1 += hv * t1[k];
        a2 += hv * t2[k];
        ae += hv * te[k];
    }
#pragma unroll
    for (int off = 16; off > 0; off >>= 1) {
        a0 += __shfl_xor(a0, off, 32);
        a1 += __shfl_xor(a1, off, 32);
        a2 += __shfl_xor(a2, off, 32);
        ae += __shfl_xor(ae, off, 32);
    }
    if (lane == 0) {
        float* o = out + (size_t)row * OUTC;
        o[0] = -1e24f;
        o[1] = ae;
        o[2] = -1e24f;
        o[3] = a0;
        o[4] = a1;
        o[5] = a2;
    }
}

extern "C" void kernel_launch(void* const* d_in, const int* in_sizes, int n_in,
                              void* d_out, int out_size, void* d_ws, size_t ws_size,
                              hipStream_t stream) {
    const int*   tokens     = (const int*)d_in[0];
    const int*   src_tokens = (const int*)d_in[1];
    const float* enc        = (const float*)d_in[2];
    const int*   enc_mask   = (const int*)d_in[3];
    const float* embed      = (const float*)d_in[4];
    const float* W1         = (const float*)d_in[5];
    const float* b1         = (const float*)d_in[6];
    const float* W2         = (const float*)d_in[7];
    const float* b2         = (const float*)d_in[8];
    float* out = (float*)d_out;

    size_t off = 0;
    auto carve = [&](size_t bytes) -> size_t {
        size_t o = off;
        off = (off + bytes + 255) & ~(size_t)255;
        return o;
    };
    char* ws = (char*)d_ws;
    __bf16* W1t  = (__bf16*)(ws + carve((size_t)HDIM * HDIM * 2));
    __bf16* W2t  = (__bf16*)(ws + carve((size_t)HDIM * HDIM * 2));
    __bf16* H1   = (__bf16*)(ws + carve((size_t)MROWS * HDIM * 2));
    __bf16* Kmat = (__bf16*)(ws + carve((size_t)MROWS * HDIM * 2));
    __bf16* Hmat = (__bf16*)(ws + carve((size_t)HROWS * HDIM * 2));
    int*    tok  = (int*)   (ws + carve((size_t)HROWS * 4));
    int*    mskA = (int*)   (ws + carve((size_t)MROWS * 4));
    (void)ws_size;

    wconv_kernel<<<(HDIM * HDIM + 255) / 256, 256, 0, stream>>>(W1, W2, W1t, W2t);
    prep_kernel<<<BATCH, 256, 0, stream>>>(tokens, src_tokens, enc_mask, tok, mskA);
    hidden_kernel<<<(HROWS * (HDIM / 8) + 255) / 256, 256, 0, stream>>>(tok, embed, Hmat);

    gemm1_kernel<<<dim3(HDIM / 32, MROWS / 32, 1), 32, 0, stream>>>(enc, W1t, b1, H1);
    gemm2_kernel<<<dim3(HDIM / 32, MROWS / 32, 1), 32, 0, stream>>>(H1, W2t, b2,
                                                                    src_tokens, embed, Kmat);
    score_kernel<<<dim3(NSRC / 64, LSEQ / 64, BATCH), 128, 0, stream>>>(Hmat, Kmat, mskA, out);
    tageos_kernel<<<(HROWS + 7) / 8, 256, 0, stream>>>(Hmat, embed, out);
}
